// RecurrentNetwork_78898549228163
// MI455X (gfx1250) — matrix operand order repaired and checked by reading, not compiled, not run on hardware
//
#include <hip/hip_runtime.h>
#include <hip/hip_bf16.h>

// ---------------------------------------------------------------------------
// Problem constants (B=2048, T=256, HID=256)
// ---------------------------------------------------------------------------
#define BB    2048
#define TT    256
#define HIDC  256
#define NGATE 1024          // 4*HID
#define KIN   96            // 70 input features padded to 3 K-chunks of 32
#define LSTM_BM 32          // batch rows per workgroup
#define GSTR  36            // gates LDS row stride (floats), padded for banks
#define HSTR  264           // h LDS row stride (bf16 halves), 528 B

typedef __attribute__((ext_vector_type(8)))  float   v8f;
typedef __attribute__((ext_vector_type(16))) __bf16  v16bf;

union FragU { v16bf v; uint4 u[2]; };

__device__ __forceinline__ unsigned short f2bf(float f) {
    unsigned int u = __builtin_bit_cast(unsigned int, f);
    unsigned int r = (u + 0x7FFFu + ((u >> 16) & 1u)) >> 16;   // RNE
    return (unsigned short)r;
}
__device__ __forceinline__ float bf2f(unsigned short h) {
    unsigned int u = ((unsigned int)h) << 16;
    return __builtin_bit_cast(float, u);
}
__device__ __forceinline__ float sigm(float x) {
    return 1.0f / (1.0f + __expf(-x));
}

// Load a B-fragment: 32 contiguous bytes per lane (pre-swizzled blocks).
__device__ __forceinline__ v16bf loadB(const unsigned short* p) {
    FragU f;
    const uint4* q = (const uint4*)p;
    f.u[0] = q[0];
    f.u[1] = q[1];
    return f.v;
}
// Load an A-fragment: two 16-byte runs; per-lane halves K+{0..7},K+{16..23}.
__device__ __forceinline__ v16bf loadA(const unsigned short* p) {
    FragU f;
    f.u[0] = *(const uint4*)p;
    f.u[1] = *(const uint4*)(p + 16);
    return f.v;
}

// ---------------------------------------------------------------------------
// Kernel 1: conv(2x2,4ch) + ReLU + one-hot + reward/done -> bf16 feat [B*T][96]
// ---------------------------------------------------------------------------
__global__ void featurize_kernel(const float* __restrict__ x,
                                 const int*   __restrict__ pa,
                                 const float* __restrict__ pr,
                                 const float* __restrict__ pd,
                                 const float* __restrict__ cw,
                                 const float* __restrict__ cb,
                                 unsigned short* __restrict__ feat) {
    int idx = blockIdx.x * blockDim.x + threadIdx.x;
    if (idx >= BB * TT * KIN) return;
    int bt = idx / KIN;
    int j  = idx - bt * KIN;
    float v = 0.0f;
    if (j < 64) {
        int oc = j >> 4, oy = (j >> 2) & 3, ox = j & 3;
        float s = cb[oc];
        const float* xb = x + (size_t)bt * 50;       // (5,5,2) HWC
        #pragma unroll
        for (int ic = 0; ic < 2; ++ic)
            #pragma unroll
            for (int ky = 0; ky < 2; ++ky)
                #pragma unroll
                for (int kx = 0; kx < 2; ++kx)
                    s += xb[(oy + ky) * 10 + (ox + kx) * 2 + ic] *
                         cw[oc * 8 + ic * 4 + ky * 2 + kx];
        v = fmaxf(s, 0.0f);
    } else if (j < 68) {
        v = (pa[bt] == (j - 64)) ? 1.0f : 0.0f;
    } else if (j == 68) {
        v = pr[bt];
    } else if (j == 69) {
        v = pd[bt];
    }
    feat[idx] = f2bf(v);
}

// ---------------------------------------------------------------------------
// Kernel 2: swizzle w_ih (1024x70 -> pad 96) and w_hh (1024x256) into WMMA
// B-fragment blocks (per (ntile,kchunk): 32 lanes x 32 B), plus bias = b_ih+b_hh.
// B-frag layout (16-bit, 32x16): lane L elem j -> W[n][k],
//   n = nt*16 + (L&15), k = kc*32 + (L>>4)*8 + (j&7) + (j>>3)*16
// ---------------------------------------------------------------------------
__global__ void swizzle_kernel(const float* __restrict__ wih,
                               const float* __restrict__ whh,
                               const float* __restrict__ bih,
                               const float* __restrict__ bhh,
                               unsigned short* __restrict__ wihf,
                               unsigned short* __restrict__ whhf,
                               float* __restrict__ bias) {
    const int NIH = 64 * 3 * 512;   // 98304 halves
    const int NHH = 64 * 8 * 512;   // 262144 halves
    int idx = blockIdx.x * blockDim.x + threadIdx.x;
    if (idx < NIH) {
        int block = idx >> 9;
        int r = idx & 511, L = r >> 4, j = r & 15;
        int nt = block / 3, kc = block % 3;
        int n = nt * 16 + (L & 15);
        int k = kc * 32 + ((L >> 4) << 3) + (j & 7) + ((j >> 3) << 4);
        float v = (k < 70) ? wih[n * 70 + k] : 0.0f;
        wihf[idx] = f2bf(v);
    } else if (idx < NIH + NHH) {
        int id2 = idx - NIH;
        int block = id2 >> 9;
        int r = id2 & 511, L = r >> 4, j = r & 15;
        int nt = block >> 3, kc = block & 7;
        int n = nt * 16 + (L & 15);
        int k = kc * 32 + ((L >> 4) << 3) + (j & 7) + ((j >> 3) << 4);
        whhf[id2] = f2bf(whh[n * 256 + k]);
    } else if (idx < NIH + NHH + NGATE) {
        int n = idx - (NIH + NHH);
        bias[n] = bih[n] + bhh[n];
    }
}

// ---------------------------------------------------------------------------
// Kernel 3: fused LSTM. One WG = 32 batch rows, 8 waves, wave w owns gate
// columns [w*128, w*128+128). Per step: gates = feat_t@Wih^T + h@Whh^T + bias
// via v_wmma_f32_16x16x32_bf16, gates staged transposed in LDS, then the
// elementwise update (thread = hid column; c state in LDS).
// Weight fragments are deliberately re-loaded from L2 every step: a laundered
// zero *offset* (not pointer -> keeps global addrspace provenance, so these
// stay global_load_b128) blocks LICM from hoisting + spilling them.
// ---------------------------------------------------------------------------
__global__ void __launch_bounds__(256, 1)
lstm_kernel(const unsigned short* __restrict__ feat,
            const unsigned short* __restrict__ wihf,
            const unsigned short* __restrict__ whhf,
            const float* __restrict__ bias,
            const float* __restrict__ hidden_in,
            float* __restrict__ features_out,
            float* __restrict__ hidden_out) {
    extern __shared__ char smem[];
    float*          gates = (float*)smem;                                   // [1024][GSTR]
    unsigned short* hbuf  = (unsigned short*)(smem + NGATE * GSTR * 4);     // [32][HSTR]
    float*          cbuf  = (float*)(smem + NGATE * GSTR * 4
                                          + LSTM_BM * HSTR * 2);            // [32][256]

    const int tid    = threadIdx.x;           // 0..255 ; doubles as hid index
    const int wave   = tid >> 5;
    const int lane   = tid & 31;
    const int lane16 = lane & 15;
    const int sel    = (lane >> 4) << 3;      // 0 or 8 (A-frag half select)
    const int b0     = blockIdx.x * LSTM_BM;

    // init: h -> LDS (bf16), c -> LDS (f32)
    #pragma unroll
    for (int m = 0; m < LSTM_BM; ++m) {
        float hv = hidden_in[(size_t)(b0 + m) * 512 + tid];
        cbuf[m * HIDC + tid] = hidden_in[(size_t)(b0 + m) * 512 + 256 + tid];
        hbuf[m * HSTR + tid] = f2bf(hv);
    }

    // t-invariant per-wave bases (keeps inner loop to base+immediate addressing)
    const unsigned short* wbih = wihf + ((wave * 24) << 9) + (lane << 4); // nt*3+kc blocks
    const unsigned short* wbhh = whhf + ((wave * 64) << 9) + (lane << 4); // nt*8+kc blocks
    const unsigned short* h0   = hbuf + lane16 * HSTR + sel;
    const unsigned short* h1   = h0 + 16 * HSTR;
    float*                g0   = gates + (wave * 128 + lane16) * GSTR + sel;
    const unsigned short* f0   = feat + ((size_t)(b0 + lane16) * TT) * KIN + sel;
    const unsigned short* f1   = f0 + (size_t)16 * TT * KIN;

    // bias values for this wave's 8 N-tiles (t-invariant)
    float bv[8];
    #pragma unroll
    for (int nt = 0; nt < 8; ++nt)
        bv[nt] = bias[wave * 128 + nt * 16 + lane16];

    __syncthreads();

    for (int t = 0; t < TT; ++t) {
        // Opaque zero offset, redefined every iteration: the weight-fragment
        // loads depend on it, so LICM cannot hoist them out of the t-loop
        // (which previously caused scratch spilling). Base pointers keep
        // their global provenance -> global_load_b128, not flat_load.
        unsigned int toff = 0;
        asm volatile("" : "+v"(toff));
        const unsigned short* wih_t = wbih + toff;
        const unsigned short* whh_t = wbhh + toff;

        v8f acc[2][8];
        const v8f vzero = {0.f, 0.f, 0.f, 0.f, 0.f, 0.f, 0.f, 0.f};
        #pragma unroll
        for (int mt = 0; mt < 2; ++mt)
            #pragma unroll
            for (int nt = 0; nt < 8; ++nt)
                acc[mt][nt] = vzero;

        // ---- input contribution: K = 3 chunks of 32 (feat padded to 96) ----
        const unsigned short* ft0 = f0 + t * KIN;
        const unsigned short* ft1 = f1 + t * KIN;
        #pragma unroll
        for (int kc = 0; kc < 3; ++kc) {
            v16bf a0 = loadA(ft0 + kc * 32);
            v16bf a1 = loadA(ft1 + kc * 32);
            #pragma unroll
            for (int nt = 0; nt < 8; ++nt) {
                v16bf b = loadB(wih_t + ((nt * 3 + kc) << 9));
                acc[0][nt] = __builtin_amdgcn_wmma_f32_16x16x32_bf16(
                    false, a0, false, b, (short)0, acc[0][nt], false, false);
                acc[1][nt] = __builtin_amdgcn_wmma_f32_16x16x32_bf16(
                    false, a1, false, b, (short)0, acc[1][nt], false, false);
            }
        }
        // ---- hidden contribution: K = 8 chunks of 32 (h bf16 in LDS) ----
        #pragma unroll
        for (int kc = 0; kc < 8; ++kc) {
            v16bf a0 = loadA(h0 + kc * 32);
            v16bf a1 = loadA(h1 + kc * 32);
            #pragma unroll
            for (int nt = 0; nt < 8; ++nt) {
                v16bf b = loadB(whh_t + ((nt * 8 + kc) << 9));
                acc[0][nt] = __builtin_amdgcn_wmma_f32_16x16x32_bf16(
                    false, a0, false, b, (short)0, acc[0][nt], false, false);
                acc[1][nt] = __builtin_amdgcn_wmma_f32_16x16x32_bf16(
                    false, a1, false, b, (short)0, acc[1][nt], false, false);
            }
        }

        // ---- store gates (+bias) transposed [n][m] into LDS ----
        // D-frag: elem r of lane L -> (M = mt*16 + r + (L>=16?8:0), N = nt*16 + L%16)
        #pragma unroll
        for (int mt = 0; mt < 2; ++mt) {
            #pragma unroll
            for (int nt = 0; nt < 8; ++nt) {
                float* dst = g0 + nt * 16 * GSTR + mt * 16;
                v8f a = acc[mt][nt];
                float4 lo = {a[0] + bv[nt], a[1] + bv[nt], a[2] + bv[nt], a[3] + bv[nt]};
                float4 hi = {a[4] + bv[nt], a[5] + bv[nt], a[6] + bv[nt], a[7] + bv[nt]};
                *(float4*)dst       = lo;
                *(float4*)(dst + 4) = hi;
            }
        }
        __syncthreads();

        // ---- elementwise LSTM update; thread = hid column (coalesced stores) ----
        #pragma unroll
        for (int m = 0; m < LSTM_BM; ++m) {
            float gi = gates[(tid      ) * GSTR + m];
            float gf = gates[(tid + 256) * GSTR + m];
            float gg = gates[(tid + 512) * GSTR + m];
            float go = gates[(tid + 768) * GSTR + m];
            float cv = sigm(gf) * cbuf[m * HIDC + tid] + sigm(gi) * tanhf(gg);
            cbuf[m * HIDC + tid] = cv;
            float hv = sigm(go) * tanhf(cv);
            hbuf[m * HSTR + tid] = f2bf(hv);
            features_out[((size_t)(b0 + m) * TT + t) * HIDC + tid] = hv;
        }
        __syncthreads();
    }

    // ---- final hidden state: concat(h, c) ----
    #pragma unroll
    for (int m = 0; m < LSTM_BM; ++m) {
        hidden_out[(size_t)(b0 + m) * 512 + tid]       = bf2f(hbuf[m * HSTR + tid]);
        hidden_out[(size_t)(b0 + m) * 512 + 256 + tid] = cbuf[m * HIDC + tid];
    }
}

// ---------------------------------------------------------------------------
// Host launcher
// ---------------------------------------------------------------------------
extern "C" void kernel_launch(void* const* d_in, const int* in_sizes, int n_in,
                              void* d_out, int out_size, void* d_ws, size_t ws_size,
                              hipStream_t stream) {
    (void)in_sizes; (void)n_in; (void)out_size; (void)ws_size;

    const float* x      = (const float*)d_in[0];
    const float* hidden = (const float*)d_in[1];
    const int*   pa     = (const int*)  d_in[2];
    const float* pr     = (const float*)d_in[3];
    const float* pd     = (const float*)d_in[4];
    const float* cw     = (const float*)d_in[5];
    const float* cb     = (const float*)d_in[6];
    const float* wih    = (const float*)d_in[7];
    const float* whh    = (const float*)d_in[8];
    const float* bih    = (const float*)d_in[9];
    const float* bhh    = (const float*)d_in[10];

    // workspace layout
    char* ws = (char*)d_ws;
    float*          bias = (float*)ws;                               // 4 KB
    unsigned short* wihf = (unsigned short*)(ws + 4096);             // 192 KB
    unsigned short* whhf = (unsigned short*)(ws + 4096 + 196608);    // 512 KB
    unsigned short* feat = (unsigned short*)(ws + (1 << 20));        // ~96 MB

    float* features_out = (float*)d_out;                             // (B,T,HID)
    float* hidden_out   = features_out + (size_t)BB * TT * HIDC;     // (B,2*HID)

    // 1) featurize: B*T*96 elements
    featurize_kernel<<<(BB * TT * KIN) / 256, 256, 0, stream>>>(
        x, pa, pr, pd, cw, cb, feat);

    // 2) weight swizzle + bias fold: 98304 + 262144 + 1024 = 361472 threads
    swizzle_kernel<<<361472 / 256, 256, 0, stream>>>(
        wih, whh, bih, bhh, wihf, whhf, bias);

    // 3) fused recurrent kernel: 64 WGs x 256 threads, ~192.5 KB dynamic LDS
    size_t smem = (size_t)NGATE * GSTR * 4 + (size_t)LSTM_BM * HSTR * 2
                + (size_t)LSTM_BM * HIDC * 4;
    lstm_kernel<<<BB / LSTM_BM, 256, smem, stream>>>(
        feat, wihf, whhf, bias, hidden, features_out, hidden_out);
}
